// AxialLinearAttention_69698729279713
// MI455X (gfx1250) — matrix operand order repaired and checked by reading, not compiled, not run on hardware
//
#include <hip/hip_runtime.h>
#include <hip/hip_bf16.h>

// ---------------------------------------------------------------------------
// Types for CDNA5 WMMA (wave32): v_wmma_f32_16x16x32_bf16
// ---------------------------------------------------------------------------
typedef __attribute__((ext_vector_type(16))) __bf16 v16bf;
typedef __attribute__((ext_vector_type(8)))  float  v8f;
typedef __attribute__((ext_vector_type(8)))  __bf16 bf16x8;
typedef __attribute__((ext_vector_type(4)))  __bf16 bf16x4;

__device__ __forceinline__ v8f wmma_bf16(v16bf a, v16bf b, v8f c) {
  // (neg_a, A, neg_b, B, c_mod, C, reuse_a, reuse_b)
  return __builtin_amdgcn_wmma_f32_16x16x32_bf16(false, a, false, b, (short)0, c,
                                                 false, false);
}

// A-matrix 16x32 bf16 fragment (ISA 7.12.2): lanes 0-15 hold K=0..7 (V0-3) and
// K=16..23 (V4-7); lanes 16-31 hold K=8..15 and K=24..31.  p = row base.
__device__ __forceinline__ v16bf load_fragA(const __bf16* p, int lh) {
  union { v16bf v; bf16x8 h[2]; } u;
  u.h[0] = *(const bf16x8*)(p + lh * 8);
  u.h[1] = *(const bf16x8*)(p + 16 + lh * 8);
  return u.v;
}

// B-matrix 32x16 bf16 fragment: lane holds column n = lane%16; lanes 0-15 hold
// K=0..7 and K=16..23, lanes 16-31 hold K=8..15 and K=24..31.  p = row of B^T.
__device__ __forceinline__ v16bf load_fragB(const __bf16* p, int lh) {
  union { v16bf v; bf16x8 h[2]; } u;
  u.h[0] = *(const bf16x8*)(p + lh * 8);
  u.h[1] = *(const bf16x8*)(p + 16 + lh * 8);
  return u.v;
}

// Generic LDS pointer -> LDS byte offset (low 32 bits of the flat address).
__device__ __forceinline__ uint32_t lds_off(const void* p) {
  return (uint32_t)(uintptr_t)p;
}

// CDNA5 async global -> LDS copy, tracked by ASYNCcnt (per-lane 16B).
__device__ __forceinline__ void async_g2l_b128(uint32_t lds, const __bf16* g) {
  asm volatile("global_load_async_to_lds_b128 %0, %1, off"
               :: "v"(lds), "v"(g) : "memory");
}
__device__ __forceinline__ void wait_async_le4() {
  asm volatile("s_wait_asynccnt 0x4" ::: "memory");
}
__device__ __forceinline__ void wait_async_0() {
  asm volatile("s_wait_asynccnt 0x0" ::: "memory");
}

// ---------------------------------------------------------------------------
// Elementwise fp32 -> bf16 convert (4 elems/thread)
// ---------------------------------------------------------------------------
__global__ void cvt_bf16_k(const float* __restrict__ in, __bf16* __restrict__ out) {
  size_t i = ((size_t)blockIdx.x * blockDim.x + threadIdx.x) * 4;
  float4 v = *(const float4*)(in + i);
  bf16x4 o;
  o.x = (__bf16)v.x; o.y = (__bf16)v.y; o.z = (__bf16)v.z; o.w = (__bf16)v.w;
  *(bf16x4*)(out + i) = o;
}

// y1 rows are (b*128 + f*32 + a); emit bf16 rows (b*32 + a)*4 + f  (grid_freq)
__global__ void permute_cvt_k(const float* __restrict__ y1, __bf16* __restrict__ out) {
  size_t oid = (size_t)blockIdx.x * blockDim.x + threadIdx.x;  // one float4 chunk
  int d4 = (int)(oid & 255);
  int r2 = (int)(oid >> 8);
  int f = r2 & 3, a = (r2 >> 2) & 31, b = r2 >> 7;
  int r1 = (b << 7) + (f << 5) + a;
  float4 v = *(const float4*)(y1 + (size_t)r1 * 1024 + d4 * 4);
  bf16x4 o;
  o.x = (__bf16)v.x; o.y = (__bf16)v.y; o.z = (__bf16)v.z; o.w = (__bf16)v.w;
  *(bf16x4*)(out + (size_t)r2 * 1024 + d4 * 4) = o;
}

// W (KxN fp32, row major) -> Wt (NxK bf16, row major)   [1024x1024]
__global__ void wcvt_k(const float* __restrict__ W, __bf16* __restrict__ Wt) {
  __shared__ float t[32][33];
  int n0 = blockIdx.x * 32, k0 = blockIdx.y * 32;
  int tx = threadIdx.x, ty = threadIdx.y;  // 32 x 8
#pragma unroll
  for (int r = 0; r < 4; ++r)
    t[ty + r * 8][tx] = W[(size_t)(k0 + ty + r * 8) * 1024 + n0 + tx];
  __syncthreads();
#pragma unroll
  for (int r = 0; r < 4; ++r)
    Wt[(size_t)(n0 + ty + r * 8) * 1024 + k0 + tx] = (__bf16)t[tx][ty + r * 8];
}

// ---------------------------------------------------------------------------
// GEMM: C[M=32768, N=1024] = act(A_bf16 @ Wt^T + bias)  (K = 1024)
// ACT: 0 none, 1 elu1.  RES_MODE: -1 bf16 out, 0 fp32 out + residual (same row),
// 1 fp32 out + residual with freq->ant row remap applied to both.
// Block tile 128x128, BK=32, 256 threads = 8 waves, wave tile 64x32.
// Double-buffered LDS fed by async global->LDS loads (ASYNCcnt pipeline).
// ---------------------------------------------------------------------------
template <int ACT, int RES_MODE>
__global__ __launch_bounds__(256) void gemm_k(
    const __bf16* __restrict__ A, const __bf16* __restrict__ Wt,
    const float* __restrict__ bias, const float* __restrict__ res,
    float* __restrict__ outf, __bf16* __restrict__ outb) {
  __shared__ __align__(16) __bf16 As[2][128][40];  // 32 used + pad (80B rows)
  __shared__ __align__(16) __bf16 Bs[2][128][40];
  const int tid = threadIdx.x;
  const int m0 = blockIdx.y * 128;
  const int n0 = blockIdx.x * 128;
  const int lane = tid & 31, wave = tid >> 5;
  const int lr = lane & 15, lh = lane >> 4;
  const int wy = wave & 1, wx = wave >> 1;  // 2 x 4 wave grid

  // Per-thread 16B chunk coordinates: chunk0 = tid, chunk1 = tid + 256.
  const int r0 = tid >> 2, c0 = (tid & 3) * 8;  // chunk1: row r0+64, same col
  const __bf16* gA0 = A  + (size_t)(m0 + r0)      * 1024 + c0;
  const __bf16* gA1 = A  + (size_t)(m0 + r0 + 64) * 1024 + c0;
  const __bf16* gB0 = Wt + (size_t)(n0 + r0)      * 1024 + c0;
  const __bf16* gB1 = Wt + (size_t)(n0 + r0 + 64) * 1024 + c0;
  uint32_t lA0[2], lA1[2], lB0[2], lB1[2];
#pragma unroll
  for (int b = 0; b < 2; ++b) {
    lA0[b] = lds_off(&As[b][r0][c0]);
    lA1[b] = lds_off(&As[b][r0 + 64][c0]);
    lB0[b] = lds_off(&Bs[b][r0][c0]);
    lB1[b] = lds_off(&Bs[b][r0 + 64][c0]);
  }

  v8f acc[4][2] = {};

  // Prologue: fill buffer 0 (k = 0).
  async_g2l_b128(lA0[0], gA0);
  async_g2l_b128(lA1[0], gA1);
  async_g2l_b128(lB0[0], gB0);
  async_g2l_b128(lB1[0], gB1);

  for (int kt = 0; kt < 32; ++kt) {
    const int cur = kt & 1;
    if (kt < 31) {
      const int nxt = cur ^ 1;
      const int k = (kt + 1) * 32;
      async_g2l_b128(lA0[nxt], gA0 + k);
      async_g2l_b128(lA1[nxt], gA1 + k);
      async_g2l_b128(lB0[nxt], gB0 + k);
      async_g2l_b128(lB1[nxt], gB1 + k);
      wait_async_le4();  // in-order: <=4 outstanding => current buffer landed
    } else {
      wait_async_0();
    }
    __syncthreads();  // current buffer visible to all waves

    v16bf af[4], bfr[2];
#pragma unroll
    for (int i = 0; i < 4; ++i)
      af[i] = load_fragA(&As[cur][wy * 64 + i * 16 + lr][0], lh);
#pragma unroll
    for (int j = 0; j < 2; ++j)
      bfr[j] = load_fragB(&Bs[cur][wx * 32 + j * 16 + lr][0], lh);
#pragma unroll
    for (int i = 0; i < 4; ++i)
#pragma unroll
      for (int j = 0; j < 2; ++j) acc[i][j] = wmma_bf16(af[i], bfr[j], acc[i][j]);

    __syncthreads();  // reads done before this buffer is refilled at kt+2
  }

#pragma unroll
  for (int j = 0; j < 2; ++j) {
    int col = n0 + wx * 32 + j * 16 + lr;
    float bv = bias[col];
#pragma unroll
    for (int i = 0; i < 4; ++i) {
#pragma unroll
      for (int r = 0; r < 8; ++r) {
        int row = m0 + wy * 64 + i * 16 + r + lh * 8;
        float v = acc[i][j][r] + bv;
        if (ACT == 1) v = (v > 0.f) ? (v + 1.f) : __expf(v);  // elu(x)+1
        if (RES_MODE < 0) {
          outb[(size_t)row * 1024 + col] = (__bf16)v;
        } else {
          int orow = row;
          if (RES_MODE == 1) {
            int f = row & 3, a = (row >> 2) & 31, b = row >> 7;
            orow = (b << 7) + (f << 5) + a;
          }
          size_t idx = (size_t)orow * 1024 + col;
          outf[idx] = v + res[idx];
        }
      }
    }
  }
}

// ---------------------------------------------------------------------------
// Linear attention inner: per (batch, head): KV = K^T V (64x64, over L),
// O = Q @ KV (Lx64).  One 128-thread block (4 waves) per unit.
// L = 32 (ant) or 4 (freq, zero padded to 32 in LDS).
// ---------------------------------------------------------------------------
template <int L>
__global__ __launch_bounds__(128) void attn_k(
    const __bf16* __restrict__ qb, const __bf16* __restrict__ kb,
    const __bf16* __restrict__ vb, __bf16* __restrict__ ob) {
  __shared__ __align__(16) __bf16 Kt[64][40];   // K^T : d x l
  __shared__ __align__(16) __bf16 Vt[64][40];   // V^T : e x l  (B^T rows)
  __shared__ __align__(16) __bf16 Qs[32][80];   // Q   : l x d
  __shared__ __align__(16) __bf16 KVt[64][80];  // KV^T: e x d  (B^T rows)
  const int tid = threadIdx.x;
  const int unit = blockIdx.x;
  const int bg = unit >> 4, h = unit & 15;
  const int base_row = bg * L, col0 = h * 64;
  const int lane = tid & 31, w = tid >> 5;
  const int lr = lane & 15, lh = lane >> 4;

  if (L < 32) {
    for (int i = tid; i < 64 * 40; i += 128) {
      (&Kt[0][0])[i] = (__bf16)0.f;
      (&Vt[0][0])[i] = (__bf16)0.f;
    }
    for (int i = tid; i < 32 * 80; i += 128) (&Qs[0][0])[i] = (__bf16)0.f;
    __syncthreads();
  }
  for (int i = tid; i < L * 64; i += 128) {
    int l = i >> 6, d = i & 63;
    size_t g = (size_t)(base_row + l) * 1024 + col0 + d;
    Kt[d][l] = kb[g];
    Vt[d][l] = vb[g];
    Qs[l][d] = qb[g];
  }
  __syncthreads();

  // KV^T: wave w owns d-strip [16w, 16w+16); 4 WMMAs over e tiles (K-dim = 32)
  {
    v16bf ak = load_fragA(&Kt[w * 16 + lr][0], lh);
#pragma unroll
    for (int e = 0; e < 4; ++e) {
      v8f kv = {};
      v16bf bv = load_fragB(&Vt[e * 16 + lr][0], lh);
      kv = wmma_bf16(ak, bv, kv);
#pragma unroll
      for (int r = 0; r < 8; ++r) {
        int dg = w * 16 + r + lh * 8;  // C layout: M = r (+8 for hi lanes)
        int eg = e * 16 + lr;          //            N = lane % 16
        KVt[eg][dg] = (__bf16)kv[r];   // store transposed -> B^T rows
      }
    }
  }
  __syncthreads();

  // O = Q @ KV: wave w owns e-strip [16w, 16w+16); K-dim = 64 -> 2 steps
#pragma unroll
  for (int mt = 0; mt < (L + 15) / 16; ++mt) {
    v8f o = {};
#pragma unroll
    for (int ks = 0; ks < 2; ++ks) {
      v16bf aq = load_fragA(&Qs[mt * 16 + lr][ks * 32], lh);
      v16bf bk = load_fragB(&KVt[w * 16 + lr][ks * 32], lh);
      o = wmma_bf16(aq, bk, o);
    }
#pragma unroll
    for (int r = 0; r < 8; ++r) {
      int l = mt * 16 + r + lh * 8;
      if (l < L)
        ob[(size_t)(base_row + l) * 1024 + col0 + w * 16 + lr] = (__bf16)o[r];
    }
  }
}

// ---------------------------------------------------------------------------
// Host side
// ---------------------------------------------------------------------------
extern "C" void kernel_launch(void* const* d_in, const int* in_sizes, int n_in,
                              void* d_out, int out_size, void* d_ws, size_t ws_size,
                              hipStream_t stream) {
  (void)in_sizes; (void)n_in; (void)out_size; (void)ws_size;
  const float* x    = (const float*)d_in[0];
  const float* w_aq = (const float*)d_in[1];  const float* b_aq = (const float*)d_in[2];
  const float* w_ak = (const float*)d_in[3];  const float* b_ak = (const float*)d_in[4];
  const float* w_av = (const float*)d_in[5];  const float* b_av = (const float*)d_in[6];
  const float* w_ao = (const float*)d_in[7];  const float* b_ao = (const float*)d_in[8];
  const float* w_fq = (const float*)d_in[9];  const float* b_fq = (const float*)d_in[10];
  const float* w_fk = (const float*)d_in[11]; const float* b_fk = (const float*)d_in[12];
  const float* w_fv = (const float*)d_in[13]; const float* b_fv = (const float*)d_in[14];
  const float* w_fo = (const float*)d_in[15]; const float* b_fo = (const float*)d_in[16];

  const size_t MD = 32768ULL * 1024ULL;  // activations elements
  const size_t WD = 1024ULL * 1024ULL;   // weight elements
  __bf16* xb = (__bf16*)d_ws;     // also reused as grid_freq bf16
  __bf16* qb = xb + MD;
  __bf16* kb = qb + MD;
  __bf16* vb = kb + MD;
  __bf16* ob = vb + MD;
  __bf16* wt = ob + MD;           // 8 transposed bf16 weights
  float*  y1 = (float*)(wt + 8 * WD);  // ant-stage fp32 output (with residual)

  // converts
  cvt_bf16_k<<<32768, 256, 0, stream>>>(x, xb);
  const float* wsrc[8] = {w_aq, w_ak, w_av, w_ao, w_fq, w_fk, w_fv, w_fo};
  for (int i = 0; i < 8; ++i)
    wcvt_k<<<dim3(32, 32), dim3(32, 8), 0, stream>>>(wsrc[i], wt + i * WD);

  dim3 gg(8, 256), gb(256);
  // ---- antenna stage (Baug = 1024, L = 32) ----
  gemm_k<1, -1><<<gg, gb, 0, stream>>>(xb, wt + 0 * WD, b_aq, nullptr, nullptr, qb);
  gemm_k<1, -1><<<gg, gb, 0, stream>>>(xb, wt + 1 * WD, b_ak, nullptr, nullptr, kb);
  gemm_k<0, -1><<<gg, gb, 0, stream>>>(xb, wt + 2 * WD, b_av, nullptr, nullptr, vb);
  attn_k<32><<<16384, 128, 0, stream>>>(qb, kb, vb, ob);
  gemm_k<0, 0><<<gg, gb, 0, stream>>>(ob, wt + 3 * WD, b_ao, x, y1, nullptr);

  // ---- permute to freq layout ----
  permute_cvt_k<<<32768, 256, 0, stream>>>(y1, xb);

  // ---- frequency stage (Baug = 8192, L = 4) ----
  gemm_k<1, -1><<<gg, gb, 0, stream>>>(xb, wt + 4 * WD, b_fq, nullptr, nullptr, qb);
  gemm_k<1, -1><<<gg, gb, 0, stream>>>(xb, wt + 5 * WD, b_fk, nullptr, nullptr, kb);
  gemm_k<0, -1><<<gg, gb, 0, stream>>>(xb, wt + 6 * WD, b_fv, nullptr, nullptr, vb);
  attn_k<4><<<131072, 128, 0, stream>>>(qb, kb, vb, ob);
  // final out-proj: residual read and store both remapped back to (B, FG*ANT, D)
  gemm_k<0, 1><<<gg, gb, 0, stream>>>(ob, wt + 7 * WD, b_fo, y1, (float*)d_out, nullptr);
}
